// TrackHead_65326452572529
// MI455X (gfx1250) — matrix operand order repaired and checked by reading, not compiled, not run on hardware
//
#include <hip/hip_runtime.h>
#include <hip/hip_bf16.h>

typedef __attribute__((ext_vector_type(16))) __bf16 v16bf;
typedef __attribute__((ext_vector_type(8)))  __bf16 v8bf;
typedef __attribute__((ext_vector_type(8)))  float  v8f;
typedef __attribute__((ext_vector_type(4)))  float  v4f;

#define B_   2
#define T_   12
#define TM1  11
#define BT_  22
#define N_   1024
#define C_   256
#define E_   256
#define A_   11
#define H_   8
#define D_   32
#define MT   32          // tokens per workgroup in kernel 1
#define LDF  (C_ + 8)    // bf16 LDS stride for feat/token tile
#define LDY  (E_ + 4)    // f32 LDS stride for pre-LN tile
#define LN_EPSF 1e-5f
#define QSCALE  0.17677669529663687f   // 32^-0.5

__device__ __forceinline__ v8f wmma_bf16(v16bf a, v16bf b, v8f c) {
  return __builtin_amdgcn_wmma_f32_16x16x32_bf16(false, a, false, b,
                                                 (short)0, c, false, false);
}

__device__ __forceinline__ v16bf pack16(v8bf x, v8bf y) {
  v16bf r;
#pragma unroll
  for (int i = 0; i < 8; ++i) { r[i] = x[i]; r[i + 8] = y[i]; }
  return r;
}

// A tile (16 x K32, bf16): lane&15 = row; lanes 0-15 hold K{0-7,16-23},
// lanes 16-31 hold K{8-15,24-31}  (ISA 7.12.2 16-bit A layout)
__device__ __forceinline__ v16bf load_a16(const __bf16* base, int ld, int lane) {
  const __bf16* p = base + (lane & 15) * ld + (lane >> 4) * 8;
  return pack16(*(const v8bf*)p, *(const v8bf*)(p + 16));
}

// B tile (K32 x 16, bf16): lane&15 = col; lanes 0-15 K{0-7,16-23}, lanes 16-31 K{8-15,24-31}.
// Source layout: src[col][k] contiguous in k (stride ld).
__device__ __forceinline__ v16bf load_b16(const __bf16* base, int ld, int lane) {
  const __bf16* p = base + (lane & 15) * ld + (lane >> 4) * 8;
  return pack16(*(const v8bf*)p, *(const v8bf*)(p + 16));
}

// ---------------------------------------------------------------------------
// Kernel 0: one-shot fp32 -> bf16 conversion of weight matrices into ws so
// the GEMM hot loops consume bf16 directly (halves weight traffic, removes
// per-tile v_cvt chains from the WMMA loop).
// ---------------------------------------------------------------------------
__global__ __launch_bounds__(256) void k0_cvt(const float* __restrict__ src,
                                              __bf16* __restrict__ dst, int n8)
{
  const int i = blockIdx.x * 256 + threadIdx.x;
  if (i >= n8) return;
  const float* p = src + (size_t)i * 8;
  v4f a = *(const v4f*)p;
  v4f b = *(const v4f*)(p + 4);
  v8bf r;
  r[0] = (__bf16)a[0]; r[1] = (__bf16)a[1];
  r[2] = (__bf16)a[2]; r[3] = (__bf16)a[3];
  r[4] = (__bf16)b[0]; r[5] = (__bf16)b[1];
  r[6] = (__bf16)b[2]; r[7] = (__bf16)b[3];
  *(v8bf*)(dst + (size_t)i * 8) = r;
}

// ---------------------------------------------------------------------------
// Kernel 1: fused token build.  Per workgroup: 32 tokens of one set (cur/prev).
//   feat(bf16, LDS) -> WMMA feat@Wf^T (+anchor@Wa^T + bias, VALU) -> LN ->
//   WMMA tok@Wq^T / tok@Wk^T -> q/k bf16 in (bt,h,n,d) layout.
// ---------------------------------------------------------------------------
__global__ __launch_bounds__(256) void k1_tokens_qk(
    const float* __restrict__ feat, const float* __restrict__ anc,
    const float* __restrict__ Tego,
    const __bf16* __restrict__ WfB, const float* __restrict__ bfv,
    const float* __restrict__ Wa,   const float* __restrict__ ba,
    const __bf16* __restrict__ WinB, const float* __restrict__ binv,
    const float* __restrict__ gamma, const float* __restrict__ beta,
    __bf16* __restrict__ qws, __bf16* __restrict__ kws)
{
  __shared__ __bf16 sTok[MT * LDF];    // bf16 feat tile, reused for LN'd tokens
  __shared__ float  sY[MT * LDY];      // fp32 pre-LN tile
  __shared__ float  sAnc[MT * 12];
  __shared__ float  sMu[MT], sRs[MT];

  const int blk  = blockIdx.x;         // 2 * BT_ * (N_/MT) = 1408
  const int tile = blk & 31;
  const int rest = blk >> 5;
  const int bt   = rest % BT_;
  const int s    = rest / BT_;         // 0 = cur (-> q), 1 = prev (-> k)
  const int b    = bt / TM1;
  const int t    = bt % TM1;
  const int tf   = (s == 0) ? (t + 1) : t;
  const int n0   = tile * MT;

  const int tid  = threadIdx.x;
  const int lane = tid & 31;
  const int wv   = tid >> 5;

  // ---- stage features as bf16 ----
  {
    const int row = tid >> 3, c0 = (tid & 7) * 32;
    const float* fp = feat + ((size_t)(b * T_ + tf) * N_ + n0 + row) * C_ + c0;
    __bf16* dp = sTok + row * LDF + c0;
#pragma unroll
    for (int i = 0; i < 32; i += 4) {
      v4f v = *(const v4f*)(fp + i);
      dp[i]     = (__bf16)v[0]; dp[i + 1] = (__bf16)v[1];
      dp[i + 2] = (__bf16)v[2]; dp[i + 3] = (__bf16)v[3];
    }
  }
  // ---- anchors (rigid alignment for the prev set) ----
  if (tid < MT) {
    const float* ap = anc + ((size_t)(b * T_ + tf) * N_ + n0 + tid) * A_;
    float a[11], o[11];
#pragma unroll
    for (int j = 0; j < 11; ++j) a[j] = ap[j];
    if (s == 0) {
#pragma unroll
      for (int j = 0; j < 11; ++j) o[j] = a[j];
    } else {
      const float* Tm = Tego + (size_t)(b * T_ + t + 1) * 16;
      float R[3][3], tr[3];
#pragma unroll
      for (int i = 0; i < 3; ++i) {
#pragma unroll
        for (int j = 0; j < 3; ++j) R[i][j] = Tm[i * 4 + j];
        tr[i] = Tm[i * 4 + 3];
      }
#pragma unroll
      for (int i = 0; i < 3; ++i)
        o[i] = R[i][0] * a[0] + R[i][1] * a[1] + R[i][2] * a[2] + tr[i];
      o[3] = a[3]; o[4] = a[4]; o[5] = a[5];
      o[6] = R[0][0] * a[6] + R[0][1] * a[7];
      o[7] = R[1][0] * a[6] + R[1][1] * a[7];
#pragma unroll
      for (int i = 0; i < 3; ++i)
        o[8 + i] = R[i][0] * a[8] + R[i][1] * a[9] + R[i][2] * a[10];
    }
#pragma unroll
    for (int j = 0; j < 11; ++j) sAnc[tid * 12 + j] = o[j];
  }
  __syncthreads();

  const int rw = wv & 1;          // 16-row block
  const int cb = (wv >> 1) * 64;  // 64-col block
  const int hl = lane >> 4, cl = lane & 15;

  // ---- GEMM1: y = feat @ Wf^T, epilogue adds anchors @ Wa^T + biases ----
  {
    v8f acc[4];
#pragma unroll
    for (int j = 0; j < 4; ++j) acc[j] = (v8f){0, 0, 0, 0, 0, 0, 0, 0};
    for (int kk = 0; kk < C_; kk += 32) {
      v16bf aT = load_a16(sTok + rw * 16 * LDF + kk, LDF, lane);
#pragma unroll
      for (int j = 0; j < 4; ++j) {
        v16bf bT = load_b16(WfB + (size_t)(cb + j * 16) * C_ + kk, C_, lane);
        acc[j] = wmma_bf16(aT, bT, acc[j]);
      }
    }
#pragma unroll
    for (int j = 0; j < 4; ++j) {
      const int col = cb + j * 16 + cl;
      float wa[11];
#pragma unroll
      for (int jj = 0; jj < 11; ++jj) wa[jj] = Wa[col * A_ + jj];
      const float bias = bfv[col] + ba[col];
#pragma unroll
      for (int r = 0; r < 8; ++r) {
        const int row = rw * 16 + r + 8 * hl;
        float v = acc[j][r] + bias;
#pragma unroll
        for (int jj = 0; jj < 11; ++jj) v += sAnc[row * 12 + jj] * wa[jj];
        sY[row * LDY + col] = v;
      }
    }
  }
  __syncthreads();

  // ---- LayerNorm ----
  if (tid < MT) {
    float sum = 0.f, sq = 0.f;
    const float* yr = sY + tid * LDY;
    for (int c = 0; c < E_; ++c) { float v = yr[c]; sum += v; sq += v * v; }
    const float mu  = sum * (1.0f / E_);
    const float var = sq * (1.0f / E_) - mu * mu;
    sMu[tid] = mu;
    sRs[tid] = rsqrtf(var + LN_EPSF);
  }
  __syncthreads();
  {
    const int row = tid >> 3, c0 = (tid & 7) * 32;
    const float mu = sMu[row], rs = sRs[row];
#pragma unroll
    for (int i = 0; i < 32; ++i) {
      const int c = c0 + i;
      const float xn = (sY[row * LDY + c] - mu) * rs * gamma[c] + beta[c];
      sTok[row * LDF + c] = (__bf16)xn;
    }
  }
  __syncthreads();

  // ---- GEMM2: q = (tok@Wq^T + bq)*scale  /  k = tok@Wk^T + bk ----
  {
    const int wb = s * E_;   // WinB rows [0,E) = Wq, [E,2E) = Wk
    v8f acc[4];
#pragma unroll
    for (int j = 0; j < 4; ++j) acc[j] = (v8f){0, 0, 0, 0, 0, 0, 0, 0};
    for (int kk = 0; kk < E_; kk += 32) {
      v16bf aT = load_a16(sTok + rw * 16 * LDF + kk, LDF, lane);
#pragma unroll
      for (int j = 0; j < 4; ++j) {
        v16bf bT = load_b16(WinB + (size_t)(wb + cb + j * 16) * E_ + kk, E_, lane);
        acc[j] = wmma_bf16(aT, bT, acc[j]);
      }
    }
    const float scale = (s == 0) ? QSCALE : 1.0f;
    __bf16* outp = (s == 0) ? qws : kws;
#pragma unroll
    for (int j = 0; j < 4; ++j) {
      const int col = cb + j * 16 + cl;
      const float bias = binv[wb + col];
      const int h = col >> 5, d = col & 31;
#pragma unroll
      for (int r = 0; r < 8; ++r) {
        const int n = n0 + rw * 16 + r + 8 * hl;
        const float v = (acc[j][r] + bias) * scale;
        outp[(((size_t)bt * H_ + h) * N_ + n) * D_ + d] = (__bf16)v;
      }
    }
  }
}

// ---------------------------------------------------------------------------
// Kernel 2: softmax stats.  One wave = one (bt, h, 16-row n-tile); loops all
// 64 m-tiles with one WMMA each (K = D = 32), online max/sum, then merges
// across the 16-lane halves with shuffles.
// ---------------------------------------------------------------------------
__global__ __launch_bounds__(256) void k2_stats(
    const __bf16* __restrict__ qws, const __bf16* __restrict__ kws,
    float* __restrict__ Mst, float* __restrict__ Lst)
{
  const int lane = threadIdx.x & 31;
  const int task = blockIdx.x * 8 + (threadIdx.x >> 5);   // 22*8*64 tasks
  const int ntile = task & 63;
  const int h     = (task >> 6) & 7;
  const int bt    = task >> 9;

  const __bf16* qh = qws + (size_t)(bt * H_ + h) * N_ * D_;
  const __bf16* kh = kws + (size_t)(bt * H_ + h) * N_ * D_;

  v16bf aT = load_a16(qh + (size_t)ntile * 16 * D_, D_, lane);

  float mx[8], sm[8];
#pragma unroll
  for (int r = 0; r < 8; ++r) { mx[r] = -3.0e38f; sm[r] = 0.0f; }

  for (int mt = 0; mt < 64; ++mt) {
    if (mt + 8 < 64)
      __builtin_prefetch(kh + (size_t)(mt + 8) * 16 * D_ +
                         (lane & 15) * D_ + (lane >> 4) * 16, 0, 0);
    v16bf bT = load_b16(kh + (size_t)mt * 16 * D_, D_, lane);
    v8f c = (v8f){0, 0, 0, 0, 0, 0, 0, 0};
    v8f sv = wmma_bf16(aT, bT, c);
#pragma unroll
    for (int r = 0; r < 8; ++r) {
      const float s  = sv[r];
      const float nm = fmaxf(mx[r], s);
      sm[r] = sm[r] * __expf(mx[r] - nm) + __expf(s - nm);
      mx[r] = nm;
    }
  }
  // merge across the 16 lanes that share each D-matrix row
#pragma unroll
  for (int off = 1; off < 16; off <<= 1) {
#pragma unroll
    for (int r = 0; r < 8; ++r) {
      const float om = __shfl_xor(mx[r], off, 32);
      const float os = __shfl_xor(sm[r], off, 32);
      const float nm = fmaxf(mx[r], om);
      sm[r] = sm[r] * __expf(mx[r] - nm) + os * __expf(om - nm);
      mx[r] = nm;
    }
  }
  if ((lane & 15) == 0) {
    const int half = lane >> 4;
#pragma unroll
    for (int r = 0; r < 8; ++r) {
      const int n = ntile * 16 + r + 8 * half;
      const size_t idx = (size_t)(bt * H_ + h) * N_ + n;
      Mst[idx] = mx[r];
      Lst[idx] = sm[r];
    }
  }
}

// ---------------------------------------------------------------------------
// Kernel 3: attention output.  Workgroup = (bt, n-tile); 8 waves cover 8
// m-tiles of one m-group.  Each wave loops the 8 heads (one WMMA per head),
// applies exp(s-M)*inv(8L), accumulates, LDS-transposes and stores
// coalesced 32B rows to d_out.
// ---------------------------------------------------------------------------
__global__ __launch_bounds__(256) void k3_attn(
    const __bf16* __restrict__ qws, const __bf16* __restrict__ kws,
    const float* __restrict__ Mst, const float* __restrict__ Lst,
    float* __restrict__ out)
{
  __shared__ float sM[8][16];
  __shared__ float sIL[8][16];
  __shared__ float stage[8][16][20];   // rows 16B-aligned (80B stride)

  const int blk   = blockIdx.x;        // BT_*64*8 = 11264
  const int mg    = blk & 7;
  const int ntile = (blk >> 3) & 63;
  const int bt    = blk >> 9;
  const int b     = bt / TM1;
  const int t     = bt % TM1;

  const int tid  = threadIdx.x;
  const int lane = tid & 31;
  const int wv   = tid >> 5;

  if (tid < 128) {
    const int h = tid >> 4, r = tid & 15;
    sM[h][r] = Mst[(size_t)(bt * H_ + h) * N_ + ntile * 16 + r];
  } else {
    const int u = tid - 128;
    const int h = u >> 4, r = u & 15;
    sIL[h][r] = 0.125f / Lst[(size_t)(bt * H_ + h) * N_ + ntile * 16 + r];
  }
  __syncthreads();

  const int mt = mg * 8 + wv;
  const int hl = lane >> 4, cl = lane & 15;

  float acc[8];
#pragma unroll
  for (int r = 0; r < 8; ++r) acc[r] = 0.0f;

  for (int h = 0; h < H_; ++h) {
    const __bf16* qh = qws + (size_t)(bt * H_ + h) * N_ * D_;
    const __bf16* kh = kws + (size_t)(bt * H_ + h) * N_ * D_;
    v16bf aT = load_a16(qh + (size_t)ntile * 16 * D_, D_, lane);
    v16bf bT = load_b16(kh + (size_t)mt * 16 * D_, D_, lane);
    v8f c = (v8f){0, 0, 0, 0, 0, 0, 0, 0};
    v8f sv = wmma_bf16(aT, bT, c);
#pragma unroll
    for (int r = 0; r < 8; ++r) {
      const int row = r + 8 * hl;
      acc[r] += __expf(sv[r] - sM[h][row]) * sIL[h][row];
    }
  }
  // scatter accumulators into LDS tile, then write coalesced rows
#pragma unroll
  for (int r = 0; r < 8; ++r) stage[wv][r + 8 * hl][cl] = acc[r];
  __syncthreads();
  {
    const int row = lane >> 1, ch = (lane & 1) * 8;
    float* orow = out + ((size_t)(b * T_ + t + 1) * N_ + ntile * 16 + row) * N_
                      + mt * 16 + ch;
    v4f x0 = *(const v4f*)&stage[wv][row][ch];
    v4f x1 = *(const v4f*)&stage[wv][row][ch + 4];
    *(v4f*)orow       = x0;
    *(v4f*)(orow + 4) = x1;
  }
}

// ---------------------------------------------------------------------------
// Kernel 4: identity block at t = 0 for both batches.
// ---------------------------------------------------------------------------
__global__ __launch_bounds__(256) void k4_eye(float* __restrict__ out)
{
  const int b = blockIdx.x >> 10;      // 2*N_ blocks
  const int n = blockIdx.x & (N_ - 1);
  float* row = out + ((size_t)(b * T_) * N_ + n) * N_;
  const int m = threadIdx.x * 4;
  v4f v = (v4f){0, 0, 0, 0};
  if (n >= m && n < m + 4) v[n - m] = 1.0f;
  *(v4f*)(row + m) = v;
}

// ---------------------------------------------------------------------------
extern "C" void kernel_launch(void* const* d_in, const int* in_sizes, int n_in,
                              void* d_out, int out_size, void* d_ws, size_t ws_size,
                              hipStream_t stream)
{
  (void)in_sizes; (void)n_in; (void)out_size; (void)ws_size;

  const float* feat  = (const float*)d_in[0];
  const float* anc   = (const float*)d_in[1];
  const float* Tego  = (const float*)d_in[2];
  const float* Wf    = (const float*)d_in[3];
  const float* bfv   = (const float*)d_in[4];
  const float* Wa    = (const float*)d_in[5];
  const float* ba    = (const float*)d_in[6];
  const float* Win   = (const float*)d_in[7];
  const float* binv  = (const float*)d_in[8];
  const float* gamma = (const float*)d_in[9];
  const float* beta  = (const float*)d_in[10];
  float* out = (float*)d_out;

  const size_t qElems = (size_t)BT_ * H_ * N_ * D_;   // 5,767,168
  const size_t sElems = (size_t)BT_ * H_ * N_;        // 180,224
  __bf16* qws = (__bf16*)d_ws;
  __bf16* kws = qws + qElems;
  float*  Mst = (float*)(kws + qElems);
  float*  Lst = Mst + sElems;
  __bf16* WfB  = (__bf16*)(Lst + sElems);             // E_*C_ bf16
  __bf16* WinB = WfB + (size_t)E_ * C_;               // 2E_*E_ bf16
  // total ws: ~24.9 MB

  // one-shot weight conversion (fp32 -> bf16), re-run every call (stateless)
  k0_cvt<<<(E_ * C_ / 8 + 255) / 256, 256, 0, stream>>>(Wf, WfB, E_ * C_ / 8);
  k0_cvt<<<(2 * E_ * E_ / 8 + 255) / 256, 256, 0, stream>>>(Win, WinB,
                                                            2 * E_ * E_ / 8);

  k1_tokens_qk<<<2 * BT_ * (N_ / MT), 256, 0, stream>>>(
      feat, anc, Tego, WfB, bfv, Wa, ba, WinB, binv, gamma, beta, qws, kws);
  k2_stats<<<(BT_ * H_ * (N_ / 16)) / 8, 256, 0, stream>>>(qws, kws, Mst, Lst);
  k3_attn<<<BT_ * (N_ / 16) * 8, 256, 0, stream>>>(qws, kws, Mst, Lst, out);
  k4_eye<<<B_ * N_, 256, 0, stream>>>(out);
}
